// GPT_163208757555
// MI455X (gfx1250) — matrix operand order repaired and checked by reading, not compiled, not run on hardware
//
#include <hip/hip_runtime.h>
#include <hip/hip_bf16.h>
#include <math.h>

// Problem constants (match reference)
#define TT 1024
#define EE 768
#define NHEAD 12
#define HDIM 64
#define NL 4
#define BB 4
#define MMROWS (BB * TT)   // 4096 token rows
#define VV 50257

typedef __bf16 bf16_t;
typedef __attribute__((ext_vector_type(16))) __bf16 v16bf;
typedef __attribute__((ext_vector_type(8)))  float  v8f;

union FragBF {
    v16bf v;
    uint4 q[2];
};

enum { MODE_F32 = 0, MODE_GELU_BF16 = 1, MODE_RESID = 2 };

// ---------------------------------------------------------------------------
// Generic NT GEMM: C[M,N] = A[M,K] (bf16, row-major) * W[N,K]^T (bf16, row-major)
// One wave -> 32x64 C tile (2 A frags x 4 B frags, 8 WMMA per K-step of 32).
// Block = 8 waves -> 64x256 C tile.
// Ragged-N handling: B row index is CLAMPED to N-1 (no branches in the hot
// loop); columns >= N compute garbage that the epilogue never stores.
// ---------------------------------------------------------------------------
__global__ __launch_bounds__(256) void gemm_bf16_wmma(
    const bf16_t* __restrict__ A,
    const bf16_t* __restrict__ W,
    const float*  __restrict__ bias,   // [N] or nullptr
    float*        __restrict__ Cf,     // f32 out (MODE_F32 / MODE_RESID)
    bf16_t*       __restrict__ Cb,     // bf16 out (MODE_GELU_BF16)
    const float*  __restrict__ Rs,     // residual input (MODE_RESID)
    int M, int N, int K, int ldc, int mode)
{
    const int lane = threadIdx.x & 31;
    const int wave = threadIdx.x >> 5;
    const int wm   = wave >> 2;            // 0..1
    const int wn   = wave & 3;             // 0..3
    const int m0   = blockIdx.y * 64 + wm * 32;
    const int n0   = blockIdx.x * 256 + wn * 64;
    if (m0 >= M) return;

    const int g = lane >> 4;               // half-wave group
    const int l = lane & 15;

    v8f acc[2][4] = {};

    // Per-lane A bases: rows (m0+l) and (m0+16+l); K offsets {kt+g*8, kt+g*8+16}
    const bf16_t* arow0 = A + (size_t)(m0 + l) * K + g * 8;
    const bf16_t* arow1 = arow0 + (size_t)16 * K;

    // Per-lane B bases (loop-invariant, index clamped instead of branch-guarded)
    const bf16_t* brow[4];
#pragma unroll
    for (int j = 0; j < 4; ++j) {
        int n = n0 + j * 16 + l;
        n = (n < N) ? n : (N - 1);
        brow[j] = W + (size_t)n * K + g * 8;
    }

    for (int kt = 0; kt < K; kt += 32) {
        FragBF a0, a1;
        a0.q[0] = *(const uint4*)(arow0 + kt);
        a0.q[1] = *(const uint4*)(arow0 + kt + 16);
        a1.q[0] = *(const uint4*)(arow1 + kt);
        a1.q[1] = *(const uint4*)(arow1 + kt + 16);

        FragBF b[4];
#pragma unroll
        for (int j = 0; j < 4; ++j) {
            b[j].q[0] = *(const uint4*)(brow[j] + kt);
            b[j].q[1] = *(const uint4*)(brow[j] + kt + 16);
            __builtin_prefetch(brow[j] + kt + 128, 0, 1);   // global_prefetch_b8
        }
#pragma unroll
        for (int j = 0; j < 4; ++j)
            acc[0][j] = __builtin_amdgcn_wmma_f32_16x16x32_bf16(
                false, a0.v, false, b[j].v, (short)0, acc[0][j], false, false);
#pragma unroll
        for (int j = 0; j < 4; ++j)
            acc[1][j] = __builtin_amdgcn_wmma_f32_16x16x32_bf16(
                false, a1.v, false, b[j].v, (short)0, acc[1][j], false, false);
    }

    // Epilogue: element (m0 + mi*16 + r + 8g, n0 + j*16 + l) lives in acc[mi][j][r]
#pragma unroll
    for (int j = 0; j < 4; ++j) {
        int col = n0 + j * 16 + l;
        if (col >= N) continue;
        float bv = bias ? bias[col] : 0.0f;
#pragma unroll
        for (int mi = 0; mi < 2; ++mi) {
#pragma unroll
            for (int r = 0; r < 8; ++r) {
                int row  = m0 + mi * 16 + g * 8 + r;
                size_t o = (size_t)row * ldc + col;
                float v  = acc[mi][j][r] + bv;
                if (mode == MODE_F32) {
                    Cf[o] = v;
                } else if (mode == MODE_GELU_BF16) {
                    Cb[o] = (bf16_t)(0.5f * v * (1.0f + erff(v * 0.70710678f)));
                } else { // MODE_RESID
                    Cf[o] = Rs[o] + v;
                }
            }
        }
    }
}

// ---------------------------------------------------------------------------
// x[m, e] = wte[idx[m], e] + wpe[m % T, e]
// ---------------------------------------------------------------------------
__global__ __launch_bounds__(256) void embed_kernel(
    const int* __restrict__ idx, const float* __restrict__ wte,
    const float* __restrict__ wpe, float* __restrict__ x)
{
    int i = blockIdx.x * 256 + threadIdx.x;
    if (i >= MMROWS * EE) return;
    int m = i / EE;
    int e = i - m * EE;
    int t = m & (TT - 1);
    x[i] = wte[(size_t)idx[m] * EE + e] + wpe[(size_t)t * EE + e];
}

// ---------------------------------------------------------------------------
// LayerNorm: one block per row, fp32 in -> bf16 out
// ---------------------------------------------------------------------------
__global__ __launch_bounds__(256) void ln_kernel(
    const float* __restrict__ x, const float* __restrict__ w,
    const float* __restrict__ b, bf16_t* __restrict__ out)
{
    __shared__ float s1[256];
    __shared__ float s2[256];
    int row = blockIdx.x;
    const float* xr = x + (size_t)row * EE;

    float sum = 0.f, sq = 0.f;
    for (int e = threadIdx.x; e < EE; e += 256) {
        float v = xr[e];
        sum += v; sq += v * v;
    }
    s1[threadIdx.x] = sum;
    s2[threadIdx.x] = sq;
    __syncthreads();
    for (int off = 128; off > 0; off >>= 1) {
        if ((int)threadIdx.x < off) {
            s1[threadIdx.x] += s1[threadIdx.x + off];
            s2[threadIdx.x] += s2[threadIdx.x + off];
        }
        __syncthreads();
    }
    float mean = s1[0] * (1.0f / EE);
    float var  = s2[0] * (1.0f / EE) - mean * mean;
    float rstd = rsqrtf(var + 1e-5f);

    for (int e = threadIdx.x; e < EE; e += 256)
        out[(size_t)row * EE + e] = (bf16_t)((xr[e] - mean) * rstd * w[e] + b[e]);
}

// ---------------------------------------------------------------------------
// Causal attention, online softmax, chunk-of-32-keys formulation.
// One wave per (b, h, q).
//   score phase : lane jj computes s(key j0+jj); q broadcast via shuffles
//   rescale     : chunk max + exp correction amortized over 32 keys
//   value phase : lane owns head dims {2*lane, 2*lane+1}; p broadcast/key
// qkv layout: [row, 3E] = [q(768) | k(768) | v(768)], head h at offset h*64.
// ---------------------------------------------------------------------------
__global__ __launch_bounds__(128) void attn_kernel(
    const float* __restrict__ qkv, bf16_t* __restrict__ y)
{
    int w    = blockIdx.x * 4 + (threadIdx.x >> 5);
    int lane = threadIdx.x & 31;
    int q    = w & (TT - 1);
    int hb   = w >> 10;
    int h    = hb % NHEAD;
    int b    = hb / NHEAD;

    const int d0 = lane * 2;
    const size_t rowq = (size_t)(b * TT + q);
    const float* qp = qkv + rowq * (3 * EE) + h * HDIM;
    float2 qv = *(const float2*)(qp + d0);       // this lane's 2 dims of q

    const float* kbase = qkv + (size_t)(b * TT) * (3 * EE) + EE + h * HDIM;

    float mrun = -INFINITY, lsum = 0.f, a0 = 0.f, a1 = 0.f;

    for (int j0 = 0; j0 <= q; j0 += 32) {
        // ---- scores: lane computes s for key jj = j0 + lane ----
        int jj = j0 + lane;                       // always < TT (j0 <= 992)
        const float* kp = kbase + (size_t)jj * (3 * EE);
        float s = 0.f;
#pragma unroll 8
        for (int dd = 0; dd < 32; ++dd) {
            float  qa = __shfl(qv.x, dd, 32);
            float  qb = __shfl(qv.y, dd, 32);
            float2 kk = *(const float2*)(kp + 2 * dd);
            s += qa * kk.x + qb * kk.y;
        }
        s = (jj <= q) ? s * 0.125f : -INFINITY;   // 1/sqrt(64), causal mask

        // ---- chunk max + online rescale (once per 32 keys) ----
        float cmax = s;
        for (int off = 16; off > 0; off >>= 1)
            cmax = fmaxf(cmax, __shfl_xor(cmax, off, 32));
        float mn   = fmaxf(mrun, cmax);
        float corr = __expf(mrun - mn);
        float p    = __expf(s - mn);              // p == 0 for masked keys
        float psum = p;
        for (int off = 16; off > 0; off >>= 1)
            psum += __shfl_xor(psum, off, 32);
        lsum = lsum * corr + psum;
        a0 *= corr;
        a1 *= corr;
        mrun = mn;

        // ---- value accumulation: lane owns dims d0, d0+1 ----
        const float* vrow = kbase + EE + (size_t)j0 * (3 * EE) + d0;
#pragma unroll 8
        for (int t = 0; t < 32; ++t) {
            float  pt = __shfl(p, t, 32);
            float2 vv = *(const float2*)(vrow + (size_t)t * (3 * EE));
            a0 += pt * vv.x;
            a1 += pt * vv.y;
        }
    }

    float inv = 1.0f / lsum;
    bf16_t* yo = y + rowq * EE + h * HDIM;
    yo[d0]     = (bf16_t)(a0 * inv);
    yo[d0 + 1] = (bf16_t)(a1 * inv);
}

// ---------------------------------------------------------------------------
// Single-pass online log-sum-exp + NLL per row; accumulate into accum[0..1]
// ---------------------------------------------------------------------------
__global__ __launch_bounds__(256) void loss_kernel(
    const float* __restrict__ logits, const int* __restrict__ tgt,
    float* __restrict__ accum)
{
    __shared__ float shm[256];
    __shared__ float shs[256];
    int row = blockIdx.x;
    const float* lr = logits + (size_t)row * VV;

    float m = -INFINITY, s = 0.f;
    for (int i = threadIdx.x; i < VV; i += 256) {
        float v  = lr[i];
        float mn = fmaxf(m, v);
        s = s * __expf(m - mn) + __expf(v - mn);
        m = mn;
    }
    shm[threadIdx.x] = m;
    shs[threadIdx.x] = s;
    __syncthreads();
    for (int off = 128; off > 0; off >>= 1) {
        if ((int)threadIdx.x < off) {
            float m2 = shm[threadIdx.x + off];
            float s2 = shs[threadIdx.x + off];
            float m1 = shm[threadIdx.x];
            float s1 = shs[threadIdx.x];
            float mn = fmaxf(m1, m2);
            shm[threadIdx.x] = mn;
            shs[threadIdx.x] = s1 * __expf(m1 - mn) + s2 * __expf(m2 - mn);
        }
        __syncthreads();
    }
    if (threadIdx.x == 0) {
        int t = tgt[row];
        if (t != -1) {
            float nll = -(lr[t] - shm[0] - __logf(shs[0]));
            atomicAdd(&accum[0], nll);
            atomicAdd(&accum[1], 1.0f);
        }
    }
}

__global__ void zero_accum_kernel(float* accum) {
    accum[0] = 0.f;
    accum[1] = 0.f;
}

__global__ void finalize_kernel(const float* __restrict__ accum,
                                float* __restrict__ out_loss) {
    out_loss[0] = accum[0] / fmaxf(accum[1], 1.0f);
}

__global__ __launch_bounds__(256) void f32_to_bf16_kernel(
    const float* __restrict__ in, bf16_t* __restrict__ out, size_t n)
{
    size_t i = (size_t)blockIdx.x * 256 + threadIdx.x;
    size_t stride = (size_t)gridDim.x * 256;
    for (; i < n; i += stride) out[i] = (bf16_t)in[i];
}

// ---------------------------------------------------------------------------
extern "C" void kernel_launch(void* const* d_in, const int* in_sizes, int n_in,
                              void* d_out, int out_size, void* d_ws, size_t ws_size,
                              hipStream_t stream)
{
    (void)in_sizes; (void)n_in; (void)out_size; (void)ws_size;

    const int*   idx     = (const int*)d_in[0];
    const int*   targets = (const int*)d_in[1];
    const float* wte     = (const float*)d_in[2];
    const float* wpe     = (const float*)d_in[3];
    const float* ln1_w   = (const float*)d_in[4];
    const float* ln1_b   = (const float*)d_in[5];
    const float* attn_w  = (const float*)d_in[6];
    const float* attn_b  = (const float*)d_in[7];
    const float* aproj_w = (const float*)d_in[8];
    const float* aproj_b = (const float*)d_in[9];
    const float* ln2_w   = (const float*)d_in[10];
    const float* ln2_b   = (const float*)d_in[11];
    const float* fc_w    = (const float*)d_in[12];
    const float* fc_b    = (const float*)d_in[13];
    const float* proj_w  = (const float*)d_in[14];
    const float* proj_b  = (const float*)d_in[15];
    const float* lnf_w   = (const float*)d_in[16];
    const float* lnf_b   = (const float*)d_in[17];

    float* logits = (float*)d_out;

    // -------- workspace carve-up (deterministic every call) --------
    char*  ws  = (char*)d_ws;
    size_t off = 0;
    auto alloc = [&](size_t bytes) -> char* {
        char* p = ws + off;
        off = (off + bytes + 255) & ~(size_t)255;
        return p;
    };
    bf16_t* wte_bf    = (bf16_t*)alloc((size_t)VV * EE * 2);
    bf16_t* attnw_bf  = (bf16_t*)alloc((size_t)NL * 3 * EE * EE * 2);
    bf16_t* aprojw_bf = (bf16_t*)alloc((size_t)NL * EE * EE * 2);
    bf16_t* fcw_bf    = (bf16_t*)alloc((size_t)NL * 4 * EE * EE * 2);
    bf16_t* projw_bf  = (bf16_t*)alloc((size_t)NL * EE * 4 * EE * 2);
    float*  xbuf      = (float*) alloc((size_t)MMROWS * EE * 4);
    bf16_t* lnbuf     = (bf16_t*)alloc((size_t)MMROWS * EE * 2);
    float*  qkv       = (float*) alloc((size_t)MMROWS * 3 * EE * 4);
    bf16_t* atty      = (bf16_t*)alloc((size_t)MMROWS * EE * 2);
    bf16_t* hfc       = (bf16_t*)alloc((size_t)MMROWS * 4 * EE * 2);
    float*  accum     = (float*) alloc(256);

    zero_accum_kernel<<<1, 1, 0, stream>>>(accum);

    // -------- per-launch fp32 -> bf16 weight conversion --------
    f32_to_bf16_kernel<<<4096, 256, 0, stream>>>(wte,     wte_bf,    (size_t)VV * EE);
    f32_to_bf16_kernel<<<4096, 256, 0, stream>>>(attn_w,  attnw_bf,  (size_t)NL * 3 * EE * EE);
    f32_to_bf16_kernel<<<2048, 256, 0, stream>>>(aproj_w, aprojw_bf, (size_t)NL * EE * EE);
    f32_to_bf16_kernel<<<4096, 256, 0, stream>>>(fc_w,    fcw_bf,    (size_t)NL * 4 * EE * EE);
    f32_to_bf16_kernel<<<4096, 256, 0, stream>>>(proj_w,  projw_bf,  (size_t)NL * EE * 4 * EE);

    // -------- embeddings --------
    embed_kernel<<<(MMROWS * EE + 255) / 256, 256, 0, stream>>>(idx, wte, wpe, xbuf);

    const int gy = MMROWS / 64;   // 64 (block tile = 64 rows)

    // -------- transformer layers --------
    for (int i = 0; i < NL; ++i) {
        ln_kernel<<<MMROWS, 256, 0, stream>>>(xbuf, ln1_w + i * EE, ln1_b + i * EE, lnbuf);

        // qkv = ln1 @ attn_w^T + attn_b        [4096, 2304] f32
        gemm_bf16_wmma<<<dim3((3 * EE + 255) / 256, gy), 256, 0, stream>>>(
            lnbuf, attnw_bf + (size_t)i * 3 * EE * EE, attn_b + (size_t)i * 3 * EE,
            qkv, nullptr, nullptr, MMROWS, 3 * EE, EE, 3 * EE, MODE_F32);

        // causal attention -> atty (bf16)      [4096, 768]
        attn_kernel<<<(MMROWS * NHEAD) / 4, 128, 0, stream>>>(qkv, atty);

        // x += atty @ attnproj_w^T + b
        gemm_bf16_wmma<<<dim3((EE + 255) / 256, gy), 256, 0, stream>>>(
            atty, aprojw_bf + (size_t)i * EE * EE, aproj_b + (size_t)i * EE,
            xbuf, nullptr, xbuf, MMROWS, EE, EE, EE, MODE_RESID);

        ln_kernel<<<MMROWS, 256, 0, stream>>>(xbuf, ln2_w + i * EE, ln2_b + i * EE, lnbuf);

        // hfc = gelu(ln2 @ fc_w^T + b)  (bf16) [4096, 3072]
        gemm_bf16_wmma<<<dim3((4 * EE + 255) / 256, gy), 256, 0, stream>>>(
            lnbuf, fcw_bf + (size_t)i * 4 * EE * EE, fc_b + (size_t)i * 4 * EE,
            nullptr, hfc, nullptr, MMROWS, 4 * EE, EE, 4 * EE, MODE_GELU_BF16);

        // x += hfc @ proj_w^T + b   (K = 3072)
        gemm_bf16_wmma<<<dim3((EE + 255) / 256, gy), 256, 0, stream>>>(
            hfc, projw_bf + (size_t)i * EE * 4 * EE, proj_b + (size_t)i * EE,
            xbuf, nullptr, xbuf, MMROWS, EE, 4 * EE, EE, MODE_RESID);
    }

    // -------- final LN + tied LM head --------
    ln_kernel<<<MMROWS, 256, 0, stream>>>(xbuf, lnf_w, lnf_b, lnbuf);

    gemm_bf16_wmma<<<dim3((VV + 255) / 256, gy), 256, 0, stream>>>(
        lnbuf, wte_bf, nullptr, logits, nullptr, nullptr,
        MMROWS, VV, EE, VV, MODE_F32);

    // -------- loss --------
    loss_kernel<<<MMROWS, 256, 0, stream>>>(logits, targets, accum);
    finalize_kernel<<<1, 1, 0, stream>>>(accum, logits + (size_t)MMROWS * VV);
}
